// DeformableTransformerDecoderLayer_58918361366812
// MI455X (gfx1250) — matrix operand order repaired and checked by reading, not compiled, not run on hardware
//
#include <hip/hip_runtime.h>
#include <math.h>

typedef __attribute__((ext_vector_type(16))) _Float16 v16h;
typedef __attribute__((ext_vector_type(8)))  float    v8f;

#define NQc   100
#define BSc   8
#define NKc   18
#define Dc    256
#define NHc   8
#define DHc   32
#define LQc   (NQc * NKc)        // 1800
#define NTOK  (NQc * BSc * NKc)  // 14400
#define LENM  11253
#define DFFNc 1024

// ---------------------------------------------------------------------------
// Wave32 reduction helper
// ---------------------------------------------------------------------------
__device__ __forceinline__ float wred(float v) {
    #pragma unroll
    for (int m = 16; m > 0; m >>= 1) v += __shfl_xor(v, m, 32);
    return v;
}

// ---------------------------------------------------------------------------
// Elementwise add: dst = a + b   (dst may alias a)
// ---------------------------------------------------------------------------
__global__ __launch_bounds__(256)
void addk(float* dst, const float* a, const float* b, int n) {
    int i = blockIdx.x * blockDim.x + threadIdx.x;
    if (i < n) dst[i] = a[i] + b[i];
}

// ---------------------------------------------------------------------------
// Split-f16 fragment packing (done ONCE per operand, outside the GEMM loop).
// Each fp32 x -> hi = (f16)x, lo = (f16)(x - hi);  A*B ~ AhBh + AlBh + AhBl.
//
// Packed layout (per 16-row/col tile t, per K=32 step kt):
//   frag base = ((t*KT + kt)*32 + lane) * 32 halves; halves[0:16]=hi, [16:32]=lo
// so the GEMM reads each fragment with two aligned global_load_b128.
// ---------------------------------------------------------------------------

// A fragments (16-bit A layout: M = lane&15; lanes<16 K={e,e+8}, lanes>=16 K={e+8,e+16})
// aMode 1: logical row r=(b*LENM+pos) is fetched from memory stored (pos, b, D).
__global__ __launch_bounds__(256)
void pack_a_frag(const float* __restrict__ X, _Float16* __restrict__ out,
                 int M, int K, int aMode)
{
    const int lane   = threadIdx.x & 31;
    const int waveId = (blockIdx.x * blockDim.x + threadIdx.x) >> 5;
    const int Mt = (M + 15) >> 4;
    const int KT = K >> 5;
    if (waveId >= Mt * KT) return;
    const int im = waveId / KT;
    const int kt = waveId - im * KT;
    const int lm = lane & 15, hs = lane >> 4;

    const int  mrow   = im * 16 + lm;
    const bool mvalid = (mrow < M);
    long arow = mrow;
    if (aMode == 1) {
        int b = mrow / LENM;
        int p = mrow - b * LENM;
        arow = (long)p * BSc + b;
    }
    const float* Xp = X + arow * (long)K + kt * 32;

    v16h hi, lo;
    #pragma unroll
    for (int e = 0; e < 16; ++e) {
        const int kk = hs ? (e < 8 ? e + 8 : e + 16)
                          : (e < 8 ? e     : e + 8);
        const float x = mvalid ? Xp[kk] : 0.0f;
        const _Float16 h = (_Float16)x;
        hi[e] = h;
        lo[e] = (_Float16)(x - (float)h);
    }
    _Float16* o = out + ((long)waveId * 32 + lane) * 32;
    *(v16h*)(o)      = hi;
    *(v16h*)(o + 16) = lo;
}

// B fragments from W[N,K] (row-major, we need W^T): N = lane&15,
// lanes<16 supply K=kt*32+{e,e+8}, lanes>=16 supply K=kt*32+{e+8,e+16}.  N multiple of 16.
__global__ __launch_bounds__(256)
void pack_b_frag(const float* __restrict__ W, _Float16* __restrict__ out,
                 int N, int K)
{
    const int lane   = threadIdx.x & 31;
    const int waveId = (blockIdx.x * blockDim.x + threadIdx.x) >> 5;
    const int Nt16 = N >> 4;
    const int KT   = K >> 5;
    if (waveId >= Nt16 * KT) return;
    const int n16 = waveId / KT;
    const int kt  = waveId - n16 * KT;
    const int lm = lane & 15, hs = lane >> 4;

    const float* Wp = W + (long)(n16 * 16 + lm) * K + kt * 32;

    v16h hi, lo;
    #pragma unroll
    for (int e = 0; e < 16; ++e) {
        const float x = Wp[(e < 8 ? e : e + 8) + 8 * hs];
        const _Float16 h = (_Float16)x;
        hi[e] = h;
        lo[e] = (_Float16)(x - (float)h);
    }
    _Float16* o = out + ((long)waveId * 32 + lane) * 32;
    *(v16h*)(o)      = hi;
    *(v16h*)(o + 16) = lo;
}

// ---------------------------------------------------------------------------
// Packed split-f16 WMMA GEMM:  Out[M,N] = A[M,K] @ W[N,K]^T + bias[N]
// Hot loop per K=32 step: 4 b128 loads (A hi/lo) + 4x4 b128 loads (B hi/lo)
// + 12x v_wmma_f32_16x16x32_f16 + prefetch of next K tile. No conversions.
// All addresses are GEPs rooted at the kernel-arg base pointers (no pointer
// PHIs) so every access lowers to global_load_b128, keeping DScnt untouched.
// One wave per 16x64 output tile (A fragment reused across 4 B tiles).
// ---------------------------------------------------------------------------
__global__ __launch_bounds__(256)
void gemm_packed_wmma(const _Float16* __restrict__ Apack,
                      const _Float16* __restrict__ Bpack,
                      const float* __restrict__ bias, float* __restrict__ Out,
                      int M, int N, int K, int doRelu,
                      const unsigned char* __restrict__ rowMask)
{
    const int lane   = threadIdx.x & 31;
    const int waveId = (blockIdx.x * blockDim.x + threadIdx.x) >> 5;
    const int Mt = (M + 15) >> 4;
    const int Nt = N >> 6;
    const int KT = K >> 5;
    if (waveId >= Mt * Nt) return;
    const int im   = waveId % Mt;        // M-fastest: neighbor waves share B tiles
    const int in64 = waveId / Mt;
    const int m0 = im << 4;
    const int n0 = in64 << 6;
    const int hs = lane >> 4;
    const int lm = lane & 15;

    v8f acc[4];
    #pragma unroll
    for (int s = 0; s < 4; ++s) {
        const float bv = bias[n0 + s * 16 + lm];
        #pragma unroll
        for (int e = 0; e < 8; ++e) acc[s][e] = bv;
    }

    const long aBase = ((long)im * KT * 32 + lane) * 32;
    long bBase[4];
    #pragma unroll
    for (int s = 0; s < 4; ++s)
        bBase[s] = ((long)(in64 * 4 + s) * KT * 32 + lane) * 32;

    for (int kt = 0; kt < KT; ++kt) {
        const long ao = aBase + (long)kt * 1024;
        const v16h ah = *(const v16h*)(Apack + ao);
        const v16h al = *(const v16h*)(Apack + ao + 16);
        __builtin_prefetch((const void*)(Apack + ao + 1024), 0, 1);  // next A tile
        #pragma unroll
        for (int s = 0; s < 4; ++s) {
            const long bo = bBase[s] + (long)kt * 1024;
            const v16h bh = *(const v16h*)(Bpack + bo);
            const v16h bl = *(const v16h*)(Bpack + bo + 16);
            acc[s] = __builtin_amdgcn_wmma_f32_16x16x32_f16(false, ah, false, bh,
                                                            (short)0, acc[s], false, false);
            acc[s] = __builtin_amdgcn_wmma_f32_16x16x32_f16(false, al, false, bh,
                                                            (short)0, acc[s], false, false);
            acc[s] = __builtin_amdgcn_wmma_f32_16x16x32_f16(false, ah, false, bl,
                                                            (short)0, acc[s], false, false);
        }
    }

    // store (C/D layout: VGPR r -> M = m0 + r + 8*hs, N = n0 + s*16 + lane&15)
    const int mBase = m0 + (hs ? 8 : 0);
    #pragma unroll
    for (int s = 0; s < 4; ++s) {
        const int n = n0 + s * 16 + lm;
        #pragma unroll
        for (int r = 0; r < 8; ++r) {
            const int m = mBase + r;
            if (m < M) {
                float v = acc[s][r];
                if (doRelu) v = fmaxf(v, 0.0f);
                if (rowMask && rowMask[m]) v = 0.0f;
                Out[(long)m * N + n] = v;
            }
        }
    }
}

// ---------------------------------------------------------------------------
// MHA core (per (batch-item g, head h) block). qkv rows live in the natural
// token layout; token(s,g) row = s*stepS + g*stepG. Output written in the same
// layout (so projection/residual need no transpose).
//   mha1: S=18,  Bc=800, stepS=1,   stepG=18
//   mha2: S=100, Bc=144, stepS=144, stepG=1
// ---------------------------------------------------------------------------
__global__ __launch_bounds__(128)
void mha_attn(const float* __restrict__ qkv, float* __restrict__ out,
              int S, int Bc, int stepS, int stepG)
{
    const int g = blockIdx.x % Bc;
    const int h = blockIdx.x / Bc;
    __shared__ float sk[100 * DHc];
    __shared__ float sv[100 * DHc];
    const int tid = threadIdx.x;
    for (int i = tid; i < S * DHc; i += 128) {
        const int s = i >> 5, c = i & 31;
        const long row = (long)s * stepS + (long)g * stepG;
        sk[i] = qkv[row * 768 + 256 + h * DHc + c];
        sv[i] = qkv[row * 768 + 512 + h * DHc + c];
    }
    __syncthreads();
    if (tid >= S) return;
    const int  sq   = tid;
    const long qrow = (long)sq * stepS + (long)g * stepG;
    const float scale = 0.17677669529663687f;   // 1/sqrt(32)
    float q[DHc];
    #pragma unroll
    for (int c = 0; c < DHc; ++c) q[c] = qkv[qrow * 768 + h * DHc + c] * scale;

    float mx = -1e30f;
    for (int sk_i = 0; sk_i < S; ++sk_i) {
        float d = 0.f;
        #pragma unroll
        for (int c = 0; c < DHc; ++c) d += q[c] * sk[sk_i * DHc + c];
        mx = fmaxf(mx, d);
    }
    float sum = 0.f, o[DHc];
    #pragma unroll
    for (int c = 0; c < DHc; ++c) o[c] = 0.f;
    for (int sk_i = 0; sk_i < S; ++sk_i) {
        float d = 0.f;
        #pragma unroll
        for (int c = 0; c < DHc; ++c) d += q[c] * sk[sk_i * DHc + c];
        const float p = __expf(d - mx);
        sum += p;
        #pragma unroll
        for (int c = 0; c < DHc; ++c) o[c] += p * sv[sk_i * DHc + c];
    }
    const float inv = 1.f / sum;
    #pragma unroll
    for (int c = 0; c < DHc; ++c)
        out[qrow * Dc + h * DHc + c] = o[c] * inv;
}

// ---------------------------------------------------------------------------
// Residual + LayerNorm, one wave per 256-wide row.
//   mapMode 0: add rows aligned with t rows
//   mapMode 1: add rows are MSDA (b, q*18+nk) order
// ---------------------------------------------------------------------------
__global__ __launch_bounds__(256)
void residual_ln(const float* base, const float* add,
                 const float* __restrict__ gamma, const float* __restrict__ beta,
                 float* dst, int mapMode)
{
    const int wave = (blockIdx.x * blockDim.x + threadIdx.x) >> 5;
    const int lane = threadIdx.x & 31;
    if (wave >= NTOK) return;
    long arow = wave;
    if (mapMode == 1) {
        const int q_ = wave / (BSc * NKc);
        const int b_ = (wave / NKc) % BSc;
        const int nk = wave % NKc;
        arow = (long)b_ * LQc + q_ * NKc + nk;
    }
    float v[8];
    float s = 0.f;
    #pragma unroll
    for (int e = 0; e < 8; ++e) {
        const int c = e * 32 + lane;
        v[e] = base[(long)wave * Dc + c] + add[arow * Dc + c];
        s += v[e];
    }
    const float mean = wred(s) * (1.0f / Dc);
    float var = 0.f;
    #pragma unroll
    for (int e = 0; e < 8; ++e) { const float d = v[e] - mean; var += d * d; }
    var = wred(var) * (1.0f / Dc);
    const float inv = rsqrtf(var + 1e-5f);
    #pragma unroll
    for (int e = 0; e < 8; ++e) {
        const int c = e * 32 + lane;
        dst[(long)wave * Dc + c] = (v[e] - mean) * inv * gamma[c] + beta[c];
    }
}

// ---------------------------------------------------------------------------
// Q permute for MSDA: Q[(b, q*18+nk), :] = T[(q, b, nk), :]
// ---------------------------------------------------------------------------
__global__ __launch_bounds__(256)
void permute_q(const float* __restrict__ T, float* __restrict__ Q)
{
    const int r2 = blockIdx.x;     // 0..14399  (b * 1800 + j)
    const int c  = threadIdx.x;    // 0..255
    const int b  = r2 / LQc;
    const int j  = r2 - b * LQc;
    const int q_ = j / NKc;
    const int nk = j - q_ * NKc;
    const long trow = (long)(q_ * BSc + b) * NKc + nk;
    Q[(long)r2 * Dc + c] = T[trow * Dc + c];
}

// ---------------------------------------------------------------------------
// In-place softmax over contiguous groups of 16 (attention weights)
// ---------------------------------------------------------------------------
__global__ __launch_bounds__(256)
void softmax16(float* aw, int ngroups)
{
    const int i = blockIdx.x * blockDim.x + threadIdx.x;
    if (i >= ngroups) return;
    float* p = aw + (long)i * 16;
    float mx = p[0];
    #pragma unroll
    for (int e = 1; e < 16; ++e) mx = fmaxf(mx, p[e]);
    float v[16], s = 0.f;
    #pragma unroll
    for (int e = 0; e < 16; ++e) { v[e] = __expf(p[e] - mx); s += v[e]; }
    const float inv = 1.f / s;
    #pragma unroll
    for (int e = 0; e < 16; ++e) p[e] = v[e] * inv;
}

// ---------------------------------------------------------------------------
// Deformable bilinear sampling. One wave per (b, query j, head h); lane = dh
// channel -> each corner gather is one coalesced 128B line of value.
// value layout: (b, pos, h, 32). Output rows in (b, j) order.
// ---------------------------------------------------------------------------
__device__ __forceinline__ float corner_fetch(const float* __restrict__ value,
                                              int b, int ls, int H, int W,
                                              int h, int lane,
                                              int xi, int yi, float w)
{
    if (xi < 0 || xi >= W || yi < 0 || yi >= H) return 0.f;
    const long idx = ls + (long)yi * W + xi;
    return value[(((long)b * LENM + idx) * NHc + h) * DHc + lane] * w;
}

__global__ __launch_bounds__(256)
void msda_sample(const float* __restrict__ value, const float* __restrict__ off,
                 const float* __restrict__ aw, const float* __restrict__ refp,
                 float* __restrict__ outAcc)
{
    const int wave = (blockIdx.x * blockDim.x + threadIdx.x) >> 5;
    const int lane = threadIdx.x & 31;
    if (wave >= BSc * LQc * NHc) return;
    const int h  = wave % NHc;
    const int r2 = wave / NHc;            // b*1800 + j
    const int b  = r2 / LQc;
    const int j  = r2 - b * LQc;

    const int Hs[4] = {92, 46, 23, 12};
    const int Ws[4] = {92, 46, 23, 12};
    const int Ls[4] = {0, 8464, 10580, 11109};

    const float* offp = off + (long)r2 * 512 + h * 64;   // (l, p, 2)
    const float* awp  = aw  + (long)r2 * 128 + h * 16;   // (l, p)
    const float* rp   = refp + ((long)j * BSc + b) * 8;  // (l, 2)

    float acc = 0.f;
    #pragma unroll
    for (int l = 0; l < 4; ++l) {
        const int H = Hs[l], W = Ws[l], ls = Ls[l];
        const float rx = rp[l * 2 + 0], ry = rp[l * 2 + 1];
        #pragma unroll
        for (int p = 0; p < 4; ++p) {
            const float a  = awp[l * 4 + p];
            const float ox = offp[l * 8 + p * 2 + 0];
            const float oy = offp[l * 8 + p * 2 + 1];
            const float x = (rx + ox / (float)W) * W - 0.5f;   // normalizer (W,H)
            const float y = (ry + oy / (float)H) * H - 0.5f;
            const float x0f = floorf(x), y0f = floorf(y);
            const int   x0 = (int)x0f,  y0 = (int)y0f;
            const float fx = x - x0f,   fy = y - y0f;
            float sv = 0.f;
            sv += corner_fetch(value, b, ls, H, W, h, lane, x0,     y0,     (1.f - fx) * (1.f - fy));
            sv += corner_fetch(value, b, ls, H, W, h, lane, x0 + 1, y0,     fx * (1.f - fy));
            sv += corner_fetch(value, b, ls, H, W, h, lane, x0,     y0 + 1, (1.f - fx) * fy);
            sv += corner_fetch(value, b, ls, H, W, h, lane, x0 + 1, y0 + 1, fx * fy);
            acc += a * sv;
        }
    }
    outAcc[(long)r2 * Dc + h * DHc + lane] = acc;
}

// ---------------------------------------------------------------------------
// Orchestration
// ---------------------------------------------------------------------------
static inline int wave_blocks(long waves) { return (int)((waves + 7) / 8); }

static void run_gemm(const float* A, const float* W, const float* bias, float* Out,
                     int M, int N, int K, int aMode, int doRelu,
                     const unsigned char* rowMask,
                     _Float16* APACK, _Float16* WPACK, hipStream_t stream)
{
    const int Mt = (M + 15) / 16, KT = K / 32, Nt16 = N / 16;
    pack_a_frag<<<wave_blocks((long)Mt * KT), 256, 0, stream>>>(A, APACK, M, K, aMode);
    pack_b_frag<<<wave_blocks((long)Nt16 * KT), 256, 0, stream>>>(W, WPACK, N, K);
    gemm_packed_wmma<<<wave_blocks((long)Mt * (N / 64)), 256, 0, stream>>>(
        APACK, WPACK, bias, Out, M, N, K, doRelu, rowMask);
}

extern "C" void kernel_launch(void* const* d_in, const int* in_sizes, int n_in,
                              void* d_out, int out_size, void* d_ws, size_t ws_size,
                              hipStream_t stream)
{
    (void)in_sizes; (void)n_in; (void)out_size; (void)ws_size;
    const float* tgt    = (const float*)d_in[0];
    const float* pos    = (const float*)d_in[1];
    const float* refp   = (const float*)d_in[2];
    const float* mem    = (const float*)d_in[3];
    const float* w_in_w = (const float*)d_in[4];
    const float* w_in_b = (const float*)d_in[5];
    const float* w_out_w= (const float*)d_in[6];
    const float* w_out_b= (const float*)d_in[7];
    const float* a_in_w = (const float*)d_in[8];
    const float* a_in_b = (const float*)d_in[9];
    const float* a_out_w= (const float*)d_in[10];
    const float* a_out_b= (const float*)d_in[11];
    const float* off_w  = (const float*)d_in[12];
    const float* off_b  = (const float*)d_in[13];
    const float* aw_w   = (const float*)d_in[14];
    const float* aw_b   = (const float*)d_in[15];
    const float* val_w  = (const float*)d_in[16];
    const float* val_b  = (const float*)d_in[17];
    const float* outp_w = (const float*)d_in[18];
    const float* outp_b = (const float*)d_in[19];
    const float* wn_g   = (const float*)d_in[20];
    const float* wn_b   = (const float*)d_in[21];
    const float* an_g   = (const float*)d_in[22];
    const float* an_b   = (const float*)d_in[23];
    const float* n1_g   = (const float*)d_in[24];
    const float* n1_b   = (const float*)d_in[25];
    const float* n2_g   = (const float*)d_in[26];
    const float* n2_b   = (const float*)d_in[27];
    const float* l1_w   = (const float*)d_in[28];
    const float* l1_b   = (const float*)d_in[29];
    const float* l2_w   = (const float*)d_in[30];
    const float* l2_b   = (const float*)d_in[31];
    const unsigned char* padmask = (const unsigned char*)d_in[34];
    float* out = (float*)d_out;

    // ---- workspace layout (floats), with reuse ----
    float* ws   = (float*)d_ws;
    float* T    = ws;                        // 3,686,400   residual stream
    float* QKV  = T    + 3686400;            // 14,745,600  qkv / ffn-hidden
    float* ATT  = QKV  + 14745600;           // 3,686,400   attn out / msda acc
    float* PROJ = ATT  + 3686400;            // 3,686,400   projection out
    float* QBUF = PROJ + 3686400;            // 3,686,400   msda queries
    float* VAL  = QBUF + 3686400;            // 23,046,144  msda value
    _Float16* APACK = (_Float16*)(VAL + 23046144);   // 46,096,384 halves (max: value GEMM)
    _Float16* WPACK = APACK + 46096384;              // 524,288 halves
    float* OFF  = QKV;                       // reuse qkv slot during MSDA
    float* AW   = QKV + 7372800;             //  (off 7.37M + aw 1.84M < 14.75M)

    const int nElems = NTOK * Dc;            // 3,686,400

    // 1) t = tgt_pose + pos
    addk<<<(nElems + 255) / 256, 256, 0, stream>>>(T, tgt, pos, nElems);

    // 2) within-group MHA
    run_gemm(T, w_in_w, w_in_b, QKV, NTOK, 768, Dc, 0, 0, nullptr, APACK, WPACK, stream);
    mha_attn<<<800 * NHc, 128, 0, stream>>>(QKV, ATT, NKc, 800, 1, NKc);
    run_gemm(ATT, w_out_w, w_out_b, PROJ, NTOK, Dc, Dc, 0, 0, nullptr, APACK, WPACK, stream);
    residual_ln<<<(NTOK * 32 + 255) / 256, 256, 0, stream>>>(T, PROJ, wn_g, wn_b, T, 0);

    // 3) across-group MHA
    run_gemm(T, a_in_w, a_in_b, QKV, NTOK, 768, Dc, 0, 0, nullptr, APACK, WPACK, stream);
    mha_attn<<<144 * NHc, 128, 0, stream>>>(QKV, ATT, NQc, 144, 144, 1);
    run_gemm(ATT, a_out_w, a_out_b, PROJ, NTOK, Dc, Dc, 0, 0, nullptr, APACK, WPACK, stream);
    residual_ln<<<(NTOK * 32 + 255) / 256, 256, 0, stream>>>(T, PROJ, an_g, an_b, T, 0);

    // 4) MSDeformAttn
    addk<<<(nElems + 255) / 256, 256, 0, stream>>>(T, T, pos, nElems);   // t += pos
    permute_q<<<NTOK, 256, 0, stream>>>(T, QBUF);
    run_gemm(mem, val_w, val_b, VAL, BSc * LENM, Dc, Dc, 1, 0, padmask,  // value proj
             APACK, WPACK, stream);
    run_gemm(QBUF, off_w, off_b, OFF, NTOK, 512, Dc, 0, 0, nullptr, APACK, WPACK, stream);
    run_gemm(QBUF, aw_w, aw_b, AW, NTOK, 128, Dc, 0, 0, nullptr, APACK, WPACK, stream);
    softmax16<<<(NTOK * NHc + 255) / 256, 256, 0, stream>>>(AW, NTOK * NHc);
    msda_sample<<<(BSc * LQc * NHc * 32 + 255) / 256, 256, 0, stream>>>(
        VAL, OFF, AW, refp, ATT);
    run_gemm(ATT, outp_w, outp_b, PROJ, NTOK, Dc, Dc, 0, 0, nullptr, APACK, WPACK, stream);
    residual_ln<<<(NTOK * 32 + 255) / 256, 256, 0, stream>>>(T, PROJ, n1_g, n1_b, T, 1);

    // 5) FFN
    run_gemm(T, l1_w, l1_b, QKV, NTOK, DFFNc, Dc, 0, 1, nullptr, APACK, WPACK, stream);
    run_gemm(QKV, l2_w, l2_b, PROJ, NTOK, Dc, DFFNc, 0, 0, nullptr, APACK, WPACK, stream);
    residual_ln<<<(NTOK * 32 + 255) / 256, 256, 0, stream>>>(T, PROJ, n2_g, n2_b, out, 0);
}
